// IAF_64699387346975
// MI455X (gfx1250) — compile-verified, matrix operand order checked by reading, not run on hardware
//
#include <hip/hip_runtime.h>
#include <hip/hip_bf16.h>

#define T_LEN 131072
#define H     192
#define G4    768      // 4*H
#define KW    4
#define DEPTH 4
#define ROWS  64                        // timesteps per IAF tile
#define IAF_BLOCKS (T_LEN / ROWS)       // 2048

#if __has_builtin(__builtin_amdgcn_global_load_async_to_lds_b128) && \
    __has_builtin(__builtin_amdgcn_s_wait_asynccnt)
#define USE_ASYNC_LDS 1
#endif

typedef __attribute__((ext_vector_type(16))) __bf16 v16bf;
typedef __attribute__((ext_vector_type(8)))  float  v8f;
typedef int v4i __attribute__((ext_vector_type(4)));
typedef __attribute__((address_space(1))) v4i g_v4i;   // global AS pointer elem
typedef __attribute__((address_space(3))) v4i l_v4i;   // LDS AS pointer elem

union FragU { uint4 u[2]; v16bf v; };

// Load a 16x32 bf16 fragment (A layout, or B with row0 = column tile base)
// per CDNA5 ISA 7.12.2: lanes 0-15 -> k-base 0, lanes 16-31 -> k-base 8,
// dwords 0..3 = k[base..base+7], dwords 4..7 = k[base+16..base+23].
// Both 8-element runs are contiguous -> two b128 loads per lane.
__device__ inline v16bf load_frag(const __bf16* base, int row0, int k0, int lane, int ld) {
  const int half = lane >> 4;
  const int m    = lane & 15;
  const int kb   = k0 + half * 8;
  const __bf16* p = base + (size_t)(row0 + m) * ld + kb;
  FragU f;
  f.u[0] = *(const uint4*)(p);
  f.u[1] = *(const uint4*)(p + 16);
  return f.v;
}

// ---------------- prep: transpose Wh (f32), transpose+bf16 W1/W2 -------------
__global__ void prep_kernel(const float* __restrict__ Wh,
                            const float* __restrict__ w1,
                            const float* __restrict__ w2,
                            float* __restrict__ WhT,
                            __bf16* __restrict__ W1T,
                            __bf16* __restrict__ W2T) {
  int e = blockIdx.x * blockDim.x + threadIdx.x;
  if (e < H * G4) {                 // WhT[j*H + i] = Wh[i*G4 + j]
    int j = e / H, i = e % H;
    WhT[e] = Wh[i * G4 + j];
  }
  if (e < DEPTH * H * H) {          // W*T[d][n*H + k] = w*[d][k*H + n]  (bf16)
    int d = e / (H * H), r = e % (H * H);
    int n = r / H, k = r % H;
    W1T[e] = (__bf16)w1[(size_t)d * H * H + k * H + n];
    W2T[e] = (__bf16)w2[(size_t)d * H * H + k * H + n];
  }
}

// ---------------- reverse LSTM: single persistent workgroup ------------------
__global__ void __launch_bounds__(G4)
lstm_kernel(const float* __restrict__ x, const float* __restrict__ Wx,
            const float* __restrict__ WhT, const float* __restrict__ b,
            __bf16* __restrict__ hx) {
  __shared__ float h_s[H];
  __shared__ float g_s[G4];
  const int j = threadIdx.x;        // one gate column per thread (768)
  if (j < H) h_s[j] = 0.f;
  float c_reg = 0.f;
  __syncthreads();
  const float wxj = Wx[j];
  const float bj  = b[j];
  const float4* w4 = (const float4*)(WhT + (size_t)j * H);   // contiguous column
  for (int st = 0; st < T_LEN; ++st) {
    const float xt = x[T_LEN - 1 - st];                      // reversed scan
    float acc = bj + xt * wxj;
    const float4* h4 = (const float4*)h_s;
    #pragma unroll 8
    for (int i = 0; i < H / 4; ++i) {
      float4 w = w4[i], h = h4[i];
      acc += w.x * h.x + w.y * h.y + w.z * h.z + w.w * h.w;
    }
    g_s[j] = acc;
    __syncthreads();                // all reads of h_s done before overwrite
    if (j < H) {
      float ig = 1.f / (1.f + __expf(-g_s[j]));
      float fg = 1.f / (1.f + __expf(-g_s[H + j]));
      float gg = tanhf(g_s[2 * H + j]);
      float og = 1.f / (1.f + __expf(-g_s[3 * H + j]));
      c_reg = fg * c_reg + ig * gg;
      float hn = og * tanhf(c_reg);
      h_s[j] = hn;
      hx[(size_t)st * H + j] = (__bf16)hn;   // context, bf16 for WMMA reuse
    }
    __syncthreads();
  }
}

// ---------------- one IAF block (per depth): conv + dual-GEMM + update -------
__global__ void __launch_bounds__(256)
iaf_block_kernel(const float* __restrict__ z_in, float* __restrict__ z_out,
                 const __bf16* __restrict__ hx,
                 const float* __restrict__ cw, const float* __restrict__ cb,
                 const __bf16* __restrict__ W1T, const float* __restrict__ b1,
                 const __bf16* __restrict__ W2T,
                 const float* __restrict__ wmu, const float* __restrict__ bmu,
                 const float* __restrict__ wsg, const float* __restrict__ bsg,
                 float* __restrict__ partials, int depth) {
  __shared__ __align__(16) union SMem {
    struct { __bf16 c[ROWS * H]; __bf16 hx[ROWS * H]; } stage;  // 48 KB bf16
    float abuf[ROWS * H];                                       // 48 KB f32
  } sm;
  __shared__ float red[ROWS];

  const int tid = threadIdx.x;
  const int t0  = blockIdx.x * ROWS;

  // keep B matrices hot in near caches (global_prefetch_b8)
  for (int i = tid * 64; i < H * H; i += 256 * 64) {
    __builtin_prefetch(W1T + i, 0, 0);
    __builtin_prefetch(W2T + i, 0, 0);
  }

  // phase 0a: kick off hx tile -> LDS copy (async-to-LDS, tracked by ASYNCcnt,
  // overlapping with the conv VALU work below)
#ifdef USE_ASYNC_LDS
  {
    g_v4i* gsrc = (g_v4i*)(hx + (size_t)t0 * H);
    l_v4i* ldst = (l_v4i*)sm.stage.hx;
    for (int i = tid; i < ROWS * H * 2 / 16; i += 256)
      __builtin_amdgcn_global_load_async_to_lds_b128(gsrc + i, ldst + i, 0, 0);
  }
#else
  {
    const uint4* src = (const uint4*)(hx + (size_t)t0 * H);
    uint4* dst = (uint4*)sm.stage.hx;
    for (int i = tid; i < ROWS * H * 2 / 16; i += 256) dst[i] = src[i];
  }
#endif

  // phase 0b: causal conv (K=4, in-ch=1) of shifted z -> bf16 stage
  for (int e = tid; e < ROWS * H; e += 256) {
    int tl = e / H, h = e % H;
    int t  = t0 + tl;
    float acc = cb[h];
    #pragma unroll
    for (int k = 0; k < KW; ++k) {
      int idx = t + k - KW;                    // uses z[t-4..t-1], causal+shift
      float zv = (idx >= 0) ? z_in[idx] : 0.f;
      acc += cw[k * H + h] * zv;
    }
    sm.stage.c[e] = (__bf16)acc;
  }
#ifdef USE_ASYNC_LDS
  __builtin_amdgcn_s_wait_asynccnt(0);
#endif
  __syncthreads();

  // phase 1: a_tile = c@W1 + hx@W2 via v_wmma_f32_16x16x32_bf16 (shared accum).
  // A fragments are invariant across column tiles -> hoist into registers
  // (12 x v16bf = 96 VGPRs); inner loop is pure B b128 loads + WMMA.
  const int wave = tid >> 5, lane = tid & 31;
  const int rt = wave & 3;                     // 4 row tiles of 16
  v16bf a_c[6], a_h[6];
  #pragma unroll
  for (int kk = 0; kk < 6; ++kk) {
    a_c[kk] = load_frag(sm.stage.c,  rt * 16, kk * 32, lane, H);
    a_h[kk] = load_frag(sm.stage.hx, rt * 16, kk * 32, lane, H);
  }
  v8f acc[6] = {};                             // 6 column tiles per wave
  #pragma unroll
  for (int q = 0; q < 6; ++q) {
    const int nt = (wave >> 2) * 6 + q;        // 12 column tiles split over 2 waves
    #pragma unroll
    for (int kk = 0; kk < 6; ++kk) {           // K = 192 = 6 * 32
      v16bf b_1 = load_frag(W1T, nt * 16, kk * 32, lane, H);
      acc[q] = __builtin_amdgcn_wmma_f32_16x16x32_bf16(
          false, a_c[kk], false, b_1, (short)0, acc[q], false, false);
      v16bf b_2 = load_frag(W2T, nt * 16, kk * 32, lane, H);
      acc[q] = __builtin_amdgcn_wmma_f32_16x16x32_bf16(
          false, a_h[kk], false, b_2, (short)0, acc[q], false, false);
    }
  }
  __syncthreads();                             // staging dead; reuse LDS as abuf

  // phase 2: a = relu(acc + b1) -> f32 LDS (C/D layout: lane 0-15 M=j, 16-31 M=j+8)
  #pragma unroll
  for (int q = 0; q < 6; ++q) {
    const int nt  = (wave >> 2) * 6 + q;
    const int col = nt * 16 + (lane & 15);
    const float bias = b1[col];
    #pragma unroll
    for (int j = 0; j < 8; ++j) {
      int row = rt * 16 + (lane >> 4) * 8 + j;
      float v = acc[q][j] + bias;
      sm.abuf[row * H + col] = v > 0.f ? v : 0.f;
    }
  }
  __syncthreads();

  // phase 3: mu/sg projections (H->1), z update, logp partial
  if (tid < ROWS) {
    const float* arow = sm.abuf + tid * H;
    float mu = bmu[0], sg = bsg[0];
    for (int i = 0; i < H; ++i) { float a = arow[i]; mu += a * wmu[i]; sg += a * wsg[i]; }
    float sig = 1.f / (1.f + __expf(-sg));
    int t = t0 + tid;
    z_out[t] = sig * z_in[t] + (1.f - sig) * mu;
    // -log_sigmoid(sg) = softplus(-sg), numerically stable
    red[tid] = (sg > 0.f) ? log1pf(__expf(-sg)) : (-sg + log1pf(__expf(sg)));
  }
  __syncthreads();
  if (tid == 0) {
    float s = 0.f;
    for (int i = 0; i < ROWS; ++i) s += red[i];
    partials[depth * IAF_BLOCKS + blockIdx.x] = s;   // deterministic path
  }
}

// ---------------- deterministic logp reduction + output ----------------------
__global__ void reduce_logp_kernel(const float* __restrict__ partials,
                                   float* __restrict__ logp_acc) {
  __shared__ float red[256];
  int tid = threadIdx.x;
  float s = 0.f;
  for (int i = tid; i < DEPTH * IAF_BLOCKS; i += 256) s += partials[i];
  red[tid] = s; __syncthreads();
  for (int off = 128; off > 0; off >>= 1) {
    if (tid < off) red[tid] += red[tid + off];
    __syncthreads();
  }
  if (tid == 0) logp_acc[0] = red[0];
}

__global__ void finalize_kernel(const float* __restrict__ logp_acc,
                                const float* __restrict__ z_fin,
                                float* __restrict__ out) {
  int t = blockIdx.x * blockDim.x + threadIdx.x;
  if (t < T_LEN) {
    out[t]         = logp_acc[0];   // logp broadcasts one scalar over T
    out[T_LEN + t] = z_fin[t];
  }
}

extern "C" void kernel_launch(void* const* d_in, const int* in_sizes, int n_in,
                              void* d_out, int out_size, void* d_ws, size_t ws_size,
                              hipStream_t stream) {
  const float* s   = (const float*)d_in[0];
  const float* x   = (const float*)d_in[1];
  const float* Wx  = (const float*)d_in[2];
  const float* Wh  = (const float*)d_in[3];
  const float* bl  = (const float*)d_in[4];
  const float* cw  = (const float*)d_in[5];
  const float* cb  = (const float*)d_in[6];
  const float* w1  = (const float*)d_in[7];
  const float* b1  = (const float*)d_in[8];
  const float* w2  = (const float*)d_in[9];
  const float* wmu = (const float*)d_in[10];
  const float* bmu = (const float*)d_in[11];
  const float* wsg = (const float*)d_in[12];
  const float* bsg = (const float*)d_in[13];
  float* out = (float*)d_out;

  char* ws = (char*)d_ws;
  size_t off = 0;
  __bf16* hx    = (__bf16*)(ws + off); off += (size_t)T_LEN * H * 2;       // 50.3 MB
  float*  WhT   = (float*)(ws + off);  off += (size_t)G4 * H * 4;          // 576 KB
  __bf16* W1T   = (__bf16*)(ws + off); off += (size_t)DEPTH * H * H * 2;
  __bf16* W2T   = (__bf16*)(ws + off); off += (size_t)DEPTH * H * H * 2;
  float*  za    = (float*)(ws + off);  off += (size_t)T_LEN * 4;
  float*  zb    = (float*)(ws + off);  off += (size_t)T_LEN * 4;
  float*  parts = (float*)(ws + off);  off += (size_t)DEPTH * IAF_BLOCKS * 4;
  float*  lacc  = (float*)(ws + off);  off += 16;

  prep_kernel<<<(H * G4 + 255) / 256, 256, 0, stream>>>(Wh, w1, w2, WhT, W1T, W2T);
  lstm_kernel<<<1, G4, 0, stream>>>(x, Wx, WhT, bl, hx);

  const float* cur = s;                         // z starts as eps = s
  float* bufs[2] = { za, zb };
  for (int d = 0; d < DEPTH; ++d) {
    float* o = bufs[d & 1];
    iaf_block_kernel<<<IAF_BLOCKS, 256, 0, stream>>>(
        cur, o, hx,
        cw + d * KW * H, cb + d * H,
        W1T + (size_t)d * H * H, b1 + d * H,
        W2T + (size_t)d * H * H,
        wmu + d * H, bmu + d, wsg + d * H, bsg + d,
        parts, d);
    cur = o;
  }
  reduce_logp_kernel<<<1, 256, 0, stream>>>(parts, lacc);
  finalize_kernel<<<(T_LEN + 255) / 256, 256, 0, stream>>>(lacc, cur, out);
}
